// VectorizedGeometricConsistencyGraph_5274219840019
// MI455X (gfx1250) — compile-verified, matrix operand order checked by reading, not run on hardware
//
#include <hip/hip_runtime.h>
#include <hip/hip_bf16.h>

typedef __attribute__((ext_vector_type(16))) _Float16 v16h;
typedef __attribute__((ext_vector_type(8)))  _Float16 v8h;
typedef __attribute__((ext_vector_type(8)))  float    v8f;

#define BATCH 16
#define NPTS  2048
#define DIM   256
#define KG    8
#define NHEAD 4
#define HD    64

// ---------------------------------------------------------------------------
// CDNA5 async global->LDS copy (ASYNCcnt-tracked, bypasses VGPRs)
// ---------------------------------------------------------------------------
__device__ __forceinline__ void async_b128(unsigned ldsOff, const float* gsrc) {
    asm volatile("global_load_async_to_lds_b128 %0, %1, off"
                 :: "v"(ldsOff), "v"(gsrc)
                 : "memory");
}
__device__ __forceinline__ void wait_async0() {
    asm volatile("s_wait_asynccnt 0x0" ::: "memory");
}
__device__ __forceinline__ unsigned lds_off_of(const void* p) {
    // generic LDS pointer = {SHARED_BASE[63:32], lds_offset[31:0]}
    return (unsigned)(size_t)p;
}

// ---------------------------------------------------------------------------
// f32 -> f16 conversion
// ---------------------------------------------------------------------------
__global__ void cvt_f32_f16(const float* __restrict__ in, _Float16* __restrict__ out, int n) {
    int i = blockIdx.x * blockDim.x + threadIdx.x;
    if (i < n) out[i] = (_Float16)in[i];
}

// ---------------------------------------------------------------------------
// kNN + geometry features + enc MLP (writes enc straight into d_out)
// one block = 128 points of one batch; whole batch's positions + MLP weights
// staged in LDS with async b128 copies
// ---------------------------------------------------------------------------
__global__ __launch_bounds__(128) void knn_geom_enc(
    const float* __restrict__ kpt3d,
    const float* __restrict__ gw1, const float* __restrict__ gb1,
    const float* __restrict__ gw2, const float* __restrict__ gb2,
    int* __restrict__ knnIdx, float* __restrict__ geomOut,
    float* __restrict__ encOut)
{
    __shared__ float sPos[NPTS * 3];
    __shared__ float sGw1[64 * 7];
    __shared__ float sGb1[64];
    __shared__ float sGw2[DIM * 64];
    __shared__ float sGb2[DIM];

    const int b   = blockIdx.y;
    const int tid = threadIdx.x;
    const float* posB = kpt3d + (size_t)b * NPTS * 3;

    // async staging, all arrays are multiples of 16B
    {
        const unsigned oPos = lds_off_of(sPos);
        const unsigned oW1  = lds_off_of(sGw1);
        const unsigned oB1  = lds_off_of(sGb1);
        const unsigned oW2  = lds_off_of(sGw2);
        const unsigned oB2  = lds_off_of(sGb2);
        for (int t = tid; t < (NPTS * 3) / 4; t += 128) async_b128(oPos + t * 16, posB + t * 4);
        for (int t = tid; t < (64 * 7) / 4;   t += 128) async_b128(oW1  + t * 16, gw1  + t * 4);
        for (int t = tid; t < 64 / 4;         t += 128) async_b128(oB1  + t * 16, gb1  + t * 4);
        for (int t = tid; t < (DIM * 64) / 4; t += 128) async_b128(oW2  + t * 16, gw2  + t * 4);
        for (int t = tid; t < DIM / 4;        t += 128) async_b128(oB2  + t * 16, gb2  + t * 4);
        wait_async0();
    }
    __syncthreads();

    const int i = blockIdx.x * 128 + tid;
    const float px = sPos[i * 3 + 0], py = sPos[i * 3 + 1], pz = sPos[i * 3 + 2];

    float bd[KG]; int bi[KG];
#pragma unroll
    for (int s = 0; s < KG; ++s) { bd[s] = 3.0e38f; bi[s] = 0; }

    for (int j = 0; j < NPTS; ++j) {
        float dx = sPos[j * 3 + 0] - px;
        float dy = sPos[j * 3 + 1] - py;
        float dz = sPos[j * 3 + 2] - pz;
        float d2 = dx * dx + dy * dy + dz * dz;
        if (j == i) continue;
        if (d2 < bd[KG - 1]) {
            float cd = d2; int ci = j;
#pragma unroll
            for (int s = 0; s < KG; ++s) {
                bool sw = cd < bd[s];
                float td = bd[s]; int ti = bi[s];
                bd[s] = sw ? cd : td;  bi[s] = sw ? ci : ti;
                cd   = sw ? td : cd;   ci   = sw ? ti : ci;
            }
        }
    }

    // neighbor distance stats
    float nd[KG]; float mean = 0.f;
#pragma unroll
    for (int s = 0; s < KG; ++s) { nd[s] = sqrtf(bd[s]); mean += nd[s]; }
    mean *= (1.0f / KG);
    float var = 0.f;
#pragma unroll
    for (int s = 0; s < KG; ++s) { float dd = nd[s] - mean; var += dd * dd; }
    float stdv = sqrtf(var * (1.0f / (KG - 1)));

    // neighbor positions, centroid, covariance
    float nx[KG], ny[KG], nz[KG];
    float cx = 0.f, cy = 0.f, cz = 0.f;
#pragma unroll
    for (int s = 0; s < KG; ++s) {
        int j = bi[s];
        nx[s] = sPos[j * 3 + 0]; ny[s] = sPos[j * 3 + 1]; nz[s] = sPos[j * 3 + 2];
        cx += nx[s]; cy += ny[s]; cz += nz[s];
    }
    cx *= (1.0f / KG); cy *= (1.0f / KG); cz *= (1.0f / KG);
    float rx = px - cx, ry = py - cy, rz = pz - cz;

    float a00 = 0, a01 = 0, a02 = 0, a11 = 0, a12 = 0, a22 = 0;
#pragma unroll
    for (int s = 0; s < KG; ++s) {
        float ex = nx[s] - cx, ey = ny[s] - cy, ez = nz[s] - cz;
        a00 += ex * ex; a01 += ex * ey; a02 += ex * ez;
        a11 += ey * ey; a12 += ey * ez; a22 += ez * ez;
    }
    a00 *= (1.0f / KG); a01 *= (1.0f / KG); a02 *= (1.0f / KG);
    a11 *= (1.0f / KG); a12 *= (1.0f / KG); a22 *= (1.0f / KG);

    // closed-form symmetric 3x3 eigenvalues (largest / smallest)
    float q  = (a00 + a11 + a22) * (1.0f / 3.0f);
    float p1 = a01 * a01 + a02 * a02 + a12 * a12;
    float p2 = (a00 - q) * (a00 - q) + (a11 - q) * (a11 - q) + (a22 - q) * (a22 - q) + 2.0f * p1;
    float p  = sqrtf(p2 * (1.0f / 6.0f)) + 1e-20f;
    float ip = 1.0f / p;
    float b00 = (a00 - q) * ip, b11 = (a11 - q) * ip, b22 = (a22 - q) * ip;
    float b01 = a01 * ip, b02 = a02 * ip, b12 = a12 * ip;
    float detB = b00 * (b11 * b22 - b12 * b12)
               - b01 * (b01 * b22 - b12 * b02)
               + b02 * (b01 * b12 - b11 * b02);
    float r = fminf(1.0f, fmaxf(-1.0f, 0.5f * detB));
    float phi = acosf(r) * (1.0f / 3.0f);
    float e1 = q + 2.0f * p * __cosf(phi);
    float e3 = q + 2.0f * p * __cosf(phi + 2.0943951023931953f);
    float aniso = e1 / (e3 + 1e-8f);

    const size_t pointId = (size_t)b * NPTS + i;
    int*   ipOut = knnIdx  + pointId * KG;
    float* gp    = geomOut + pointId * KG * 7;
#pragma unroll
    for (int s = 0; s < KG; ++s) {
        ipOut[s] = bi[s];
        gp[s * 7 + 0] = nd[s];
        gp[s * 7 + 1] = mean;
        gp[s * 7 + 2] = stdv;
        gp[s * 7 + 3] = rx;
        gp[s * 7 + 4] = ry;
        gp[s * 7 + 5] = rz;
        gp[s * 7 + 6] = aniso;
    }

    // geom_agg -> MLP 7 -> 64 -> 256 (enc), written to output buffer
    float g[7] = { mean, mean, stdv, rx, ry, rz, aniso };
    float h[64];
#pragma unroll
    for (int j2 = 0; j2 < 64; ++j2) {
        float acc = sGb1[j2];
#pragma unroll
        for (int c = 0; c < 7; ++c) acc += sGw1[j2 * 7 + c] * g[c];
        h[j2] = fmaxf(acc, 0.0f);
    }
    float* ep = encOut + pointId * DIM;
    for (int d = 0; d < DIM; ++d) {
        float acc = sGb2[d];
#pragma unroll
        for (int j2 = 0; j2 < 64; ++j2) acc += sGw2[d * 64 + j2] * h[j2];
        ep[d] = acc;
    }
}

// ---------------------------------------------------------------------------
// WMMA GEMM: Y[m,n] = sum_k X[m,k] * W[n,k] + bias[n]
// one 16x16 output tile per wave, K in steps of 32 (v_wmma_f32_16x16x32_f16)
// A-frag layout: lanes 0-15 row m0+l, halves {K 0..7 , K 16..23}
//                lanes 16-31 row m0+l-16, halves {K 8..15, K 24..31}
// B-frag mirrors A with columns (W is row-major (Nout, K) so rows of W = cols of B)
// ---------------------------------------------------------------------------
template <bool F16OUT, bool ACCUM>
__global__ __launch_bounds__(128) void gemm_wmma_f16(
    const _Float16* __restrict__ X, const _Float16* __restrict__ W,
    const float* __restrict__ bias,
    _Float16* __restrict__ Yh, float* __restrict__ Yf,
    int Kdim, int Ndim)
{
    const int lane = threadIdx.x & 31;
    const int wave = threadIdx.x >> 5;
    const int m0   = blockIdx.x * 16;
    const int n0   = (blockIdx.y * 4 + wave) * 16;
    const int row  = m0 + (lane & 15);
    const int col  = n0 + (lane & 15);
    const int kh   = (lane >> 4) * 8;   // half-select per ISA fragment layout

    const _Float16* xp = X + (size_t)row * Kdim + kh;
    const _Float16* wp = W + (size_t)col * Kdim + kh;

    v8f c = {};
#pragma unroll
    for (int k0 = 0; k0 < 256; k0 += 32) {
        __builtin_prefetch(wp + k0 + 64, 0, 3);   // global_prefetch_b8 (near-cache)
        union { v16h v; v8h h[2]; } a, bm;
        a.h[0]  = *(const v8h*)(xp + k0);
        a.h[1]  = *(const v8h*)(xp + k0 + 16);
        bm.h[0] = *(const v8h*)(wp + k0);
        bm.h[1] = *(const v8h*)(wp + k0 + 16);
        c = __builtin_amdgcn_wmma_f32_16x16x32_f16(
                false, a.v, false, bm.v, (short)0, c, false, false);
    }

    const int   nOut  = n0 + (lane & 15);
    const int   mBase = m0 + ((lane >> 4) << 3);
    const float bb    = bias[nOut];
#pragma unroll
    for (int r = 0; r < 8; ++r) {
        float val = c[r] + bb;
        size_t o = (size_t)(mBase + r) * Ndim + nOut;
        if (F16OUT) {
            Yh[o] = (_Float16)val;
        } else {
            Yf[o] = ACCUM ? (Yf[o] + val) : val;
        }
    }
}

// ---------------------------------------------------------------------------
// graph attention: one wave per point; lane l owns channels [8l, 8l+8),
// head = lane>>3 (hd=64 -> 8 lanes per head). k=8 softmax, v8h gathers.
// ---------------------------------------------------------------------------
__global__ __launch_bounds__(256) void graph_attn(
    const _Float16* __restrict__ Qh, const _Float16* __restrict__ Kh,
    const _Float16* __restrict__ Vh,
    const int* __restrict__ knnIdx, const float* __restrict__ geom,
    const float* __restrict__ w_a, const float* __restrict__ b_a,
    _Float16* __restrict__ Aggh)
{
    const int wave  = threadIdx.x >> 5;
    const int lane  = threadIdx.x & 31;
    const int point = blockIdx.x * 8 + wave;
    const int b     = point / NPTS;
    const int head  = lane >> 3;

    const int* idx = knnIdx + (size_t)point * KG;

    v8h qv = *(const v8h*)(Qh + (size_t)point * DIM + lane * 8);
    float qf[8];
#pragma unroll
    for (int cch = 0; cch < 8; ++cch) qf[cch] = (float)qv[cch];

    float wa[7];
#pragma unroll
    for (int cch = 0; cch < 7; ++cch) wa[cch] = w_a[head * 7 + cch];
    const float ba = b_a[head];

    int   nbr[KG];
    float sc[KG];
#pragma unroll
    for (int kk = 0; kk < KG; ++kk) {
        int j = idx[kk];
        nbr[kk] = j;
        v8h kv = *(const v8h*)(Kh + ((size_t)b * NPTS + j) * DIM + lane * 8);
        float s = 0.f;
#pragma unroll
        for (int cch = 0; cch < 8; ++cch) s += qf[cch] * (float)kv[cch];
        s += __shfl_xor(s, 1, 32);
        s += __shfl_xor(s, 2, 32);
        s += __shfl_xor(s, 4, 32);
        const float* gp = geom + ((size_t)point * KG + kk) * 7;
        float bias = ba;
#pragma unroll
        for (int cch = 0; cch < 7; ++cch) bias += gp[cch] * wa[cch];
        sc[kk] = s * 0.125f + bias;    // 1/sqrt(64)
    }

    float mx = sc[0];
#pragma unroll
    for (int kk = 1; kk < KG; ++kk) mx = fmaxf(mx, sc[kk]);
    float e[KG]; float den = 0.f;
#pragma unroll
    for (int kk = 0; kk < KG; ++kk) { e[kk] = __expf(sc[kk] - mx); den += e[kk]; }
    const float inv = 1.0f / den;

    float acc[8] = {0, 0, 0, 0, 0, 0, 0, 0};
#pragma unroll
    for (int kk = 0; kk < KG; ++kk) {
        v8h vv = *(const v8h*)(Vh + ((size_t)b * NPTS + nbr[kk]) * DIM + lane * 8);
        float w = e[kk] * inv;
#pragma unroll
        for (int cch = 0; cch < 8; ++cch) acc[cch] += w * (float)vv[cch];
    }
    v8h outv;
#pragma unroll
    for (int cch = 0; cch < 8; ++cch) outv[cch] = (_Float16)acc[cch];
    *(v8h*)(Aggh + (size_t)point * DIM + lane * 8) = outv;
}

// ---------------------------------------------------------------------------
// launch
// ---------------------------------------------------------------------------
extern "C" void kernel_launch(void* const* d_in, const int* in_sizes, int n_in,
                              void* d_out, int out_size, void* d_ws, size_t ws_size,
                              hipStream_t stream) {
    const float* kpt_feature = (const float*)d_in[0];
    const float* kpt_3d      = (const float*)d_in[1];
    const float* gw1 = (const float*)d_in[2];
    const float* gb1 = (const float*)d_in[3];
    const float* gw2 = (const float*)d_in[4];
    const float* gb2 = (const float*)d_in[5];
    const float* w_q = (const float*)d_in[6];
    const float* b_q = (const float*)d_in[7];
    const float* w_k = (const float*)d_in[8];
    const float* b_k = (const float*)d_in[9];
    const float* w_v = (const float*)d_in[10];
    const float* b_v = (const float*)d_in[11];
    const float* w_a = (const float*)d_in[12];
    const float* b_a = (const float*)d_in[13];
    const float* w_o = (const float*)d_in[14];
    const float* b_o = (const float*)d_in[15];
    float* out = (float*)d_out;

    constexpr int    BN    = BATCH * NPTS;              // 32768
    constexpr size_t FEATH = (size_t)BN * DIM * 2;      // 16 MB f16 buffer
    constexpr size_t WH    = (size_t)DIM * DIM * 2;     // 128 KB f16 weight

    char* ws = (char*)d_ws;
    size_t off = 0;
    _Float16* Xh   = (_Float16*)(ws + off); off += FEATH;
    _Float16* Wqh  = (_Float16*)(ws + off); off += WH;
    _Float16* Wkh  = (_Float16*)(ws + off); off += WH;
    _Float16* Wvh  = (_Float16*)(ws + off); off += WH;
    _Float16* Woh  = (_Float16*)(ws + off); off += WH;
    _Float16* Qh   = (_Float16*)(ws + off); off += FEATH;
    _Float16* Kh   = (_Float16*)(ws + off); off += FEATH;
    _Float16* Vh   = (_Float16*)(ws + off); off += FEATH;
    _Float16* Aggh = (_Float16*)(ws + off); off += FEATH;
    int*   knnIdx  = (int*)  (ws + off);    off += (size_t)BN * KG * 4;
    float* geomBuf = (float*)(ws + off);    off += (size_t)BN * KG * 7 * 4;

    // 1) f16 staging
    cvt_f32_f16<<<(BN * DIM + 255) / 256, 256, 0, stream>>>(kpt_feature, Xh, BN * DIM);
    cvt_f32_f16<<<(DIM * DIM + 255) / 256, 256, 0, stream>>>(w_q, Wqh, DIM * DIM);
    cvt_f32_f16<<<(DIM * DIM + 255) / 256, 256, 0, stream>>>(w_k, Wkh, DIM * DIM);
    cvt_f32_f16<<<(DIM * DIM + 255) / 256, 256, 0, stream>>>(w_v, Wvh, DIM * DIM);
    cvt_f32_f16<<<(DIM * DIM + 255) / 256, 256, 0, stream>>>(w_o, Woh, DIM * DIM);

    // 2) kNN + geometry + enc (enc -> d_out)
    knn_geom_enc<<<dim3(NPTS / 128, BATCH), 128, 0, stream>>>(
        kpt_3d, gw1, gb1, gw2, gb2, knnIdx, geomBuf, out);

    // 3) Q/K/V projections (WMMA), f16 outputs
    dim3 ggrid(BN / 16, DIM / 64);
    gemm_wmma_f16<true, false><<<ggrid, 128, 0, stream>>>(Xh, Wqh, b_q, Qh, nullptr, DIM, DIM);
    gemm_wmma_f16<true, false><<<ggrid, 128, 0, stream>>>(Xh, Wkh, b_k, Kh, nullptr, DIM, DIM);
    gemm_wmma_f16<true, false><<<ggrid, 128, 0, stream>>>(Xh, Wvh, b_v, Vh, nullptr, DIM, DIM);

    // 4) graph attention -> aggregated f16 features
    graph_attn<<<BN / 8, 256, 0, stream>>>(Qh, Kh, Vh, knnIdx, geomBuf, w_a, b_a, Aggh);

    // 5) output projection, accumulated onto enc already in d_out
    gemm_wmma_f16<false, true><<<ggrid, 128, 0, stream>>>(Aggh, Woh, b_o, nullptr, out, DIM, DIM);
}
